// Graph_30837865185500
// MI455X (gfx1250) — matrix operand order repaired and checked by reading, not compiled, not run on hardware
//
#include <hip/hip_runtime.h>
#include <hip/hip_bf16.h>
#include <math.h>

// ---------------------------------------------------------------------------
// GraphSAGE (project=True, mean aggr) + graph-LayerNorm + leaky_relu, DEPTH=3
// MI455X / gfx1250: wave32, WMMA f32_16x16x32_f16 for all GEMMs.
//
// v2 changes (from disasm analysis of v1):
//  * Weights converted f32->f16 ONCE per launch (v1 re-converted the whole
//    256x256 W inside every 16-row block: ~1B v_cvt_pk VALU ops total).
//  * All WMMA-consumed activations (h, xp, agg) stored natively f16 by their
//    producing epilogues -- no precision loss vs v1 (WMMA eats f16 anyway),
//    halves GEMM A/B traffic and halves the dominant edge-gather traffic
//    (1.5 GB -> 0.75 GB, ~32us @ 23.3 TB/s). h2 stays f32 for LN stats.
//  * 32-row blocks: two A fragments share each B fragment, halving B refetch.
//    Tail tile (N=50000 % 32 != 0) clamps A-row loads (EXEC must stay all-1s
//    through WMMA) and predicates only the stores.
// ---------------------------------------------------------------------------

typedef __attribute__((ext_vector_type(16))) _Float16 v16h;
typedef __attribute__((ext_vector_type(8)))  _Float16 v8h;
typedef __attribute__((ext_vector_type(8)))  float    v8f;

#define HDIM 256
#define LN_EPS 1e-5f

__device__ __forceinline__ v16h cat8(v8h lo, v8h hi) {
  return __builtin_shufflevector(lo, hi, 0,1,2,3,4,5,6,7,8,9,10,11,12,13,14,15);
}

// ---------------------------------------------------------------------------
// WMMA GEMM:  out[r,c] = epilogue( sum_k A[r,k]*W[c,k] (+ A2@W2^T) + bias[c] )
// MODE 0: relu(A@W^T + bias) -> f16 out            (projection -> xp16)
// MODE 1: A@W^T + A2@W2^T + bias -> f32 out        (lin_l(agg) + lin_r(h))
// MODE 2: resid + A@W^T + bias -> f32 out          (final linear + residual)
// grid.x = ceil(N/32) row tiles, block = 128 threads = 4 waves; wave w owns
// cols [64w, 64w+64) for 32 rows. K = 256, swept in 8 steps of 32.
// ---------------------------------------------------------------------------
template<int MODE>
__global__ void gemm_wmma_kernel(const _Float16* __restrict__ A,
                                 const _Float16* __restrict__ W,
                                 const _Float16* __restrict__ A2,
                                 const _Float16* __restrict__ W2,
                                 const float* __restrict__ bias,
                                 const float* __restrict__ resid,
                                 void* __restrict__ outv,
                                 int nRows)
{
  const int wave   = threadIdx.x >> 5;
  const int lane   = threadIdx.x & 31;
  const int laneLo = lane & 15;
  const int laneHi = lane >> 4;
  const int row0   = blockIdx.x * 32;
  const int col0   = wave * 64;

  // Clamped A rows for the (possible) out-of-range tail tile: keeps all lanes
  // loading valid memory so EXEC stays all-1s through the WMMAs.
  const int rA0 = min(row0 + laneLo,      nRows - 1);
  const int rA1 = min(row0 + 16 + laneLo, nRows - 1);

  v8f c[2][4] = {};

  const int npass = (MODE == 1) ? 2 : 1;
  for (int pass = 0; pass < npass; ++pass) {
    const _Float16* Ap = pass ? A2 : A;
    const _Float16* Wq = pass ? W2 : W;
    const _Float16* arow0 = Ap + (size_t)rA0 * HDIM;
    const _Float16* arow1 = Ap + (size_t)rA1 * HDIM;
    #pragma unroll
    for (int ks = 0; ks < 8; ++ks) {
      const int kb = ks * 32;
      // A fragment (16-bit A 16x32, ISA 7.12.2):
      //   halves 0..7  -> K = kb + laneHi*8 + {0..7}
      //   halves 8..15 -> K = kb + 16 + laneHi*8 + {0..7}
      v16h af0, af1;
      {
        const v8h* p0 = (const v8h*)(arow0 + kb + laneHi * 8);
        const v8h* p1 = (const v8h*)(arow0 + kb + 16 + laneHi * 8);
        af0 = cat8(p0[0], p1[0]);
        const v8h* q0 = (const v8h*)(arow1 + kb + laneHi * 8);
        const v8h* q1 = (const v8h*)(arow1 + kb + 16 + laneHi * 8);
        af1 = cat8(q0[0], q1[0]);
      }
      #pragma unroll
      for (int ct = 0; ct < 4; ++ct) {
        // B fragment (B = W^T): same K order as A:
        //   halves 0..7 -> K = kb + laneHi*8 + {0..7}, 8..15 -> K = kb + 16 + laneHi*8 + {0..7}
        const _Float16* wrow = Wq + (size_t)(col0 + ct * 16 + laneLo) * HDIM
                                  + kb + laneHi * 8;
        const v8h* wp = (const v8h*)wrow;
        v16h bf = cat8(wp[0], wp[2]);
        c[0][ct] = __builtin_amdgcn_wmma_f32_16x16x32_f16(
                       false, af0, false, bf, (short)0, c[0][ct], false, false);
        c[1][ct] = __builtin_amdgcn_wmma_f32_16x16x32_f16(
                       false, af1, false, bf, (short)0, c[1][ct], false, false);
      }
    }
  }

  // Epilogue. C/D layout: c[v] = C[M = v + laneHi*8][N = laneLo].
  #pragma unroll
  for (int t = 0; t < 2; ++t) {
    #pragma unroll
    for (int ct = 0; ct < 4; ++ct) {
      const int col = col0 + ct * 16 + laneLo;
      const float bv = bias[col];
      #pragma unroll
      for (int v = 0; v < 8; ++v) {
        const int r = row0 + t * 16 + laneHi * 8 + v;
        if (r < nRows) {
          float val = c[t][ct][v] + bv;
          if (MODE == 0) {
            val = val > 0.f ? val : 0.f;
            ((_Float16*)outv)[(size_t)r * HDIM + col] = (_Float16)val;
          } else if (MODE == 1) {
            ((float*)outv)[(size_t)r * HDIM + col] = val;
          } else {
            val += resid[(size_t)r * HDIM + col];
            ((float*)outv)[(size_t)r * HDIM + col] = val;
          }
        }
      }
    }
  }
}

// --------------------------- elementwise / graph prep ----------------------

__global__ void cvt_f32_to_f16_kernel(const float* __restrict__ s,
                                      _Float16* __restrict__ d, int n)
{
  for (int i = blockIdx.x * blockDim.x + threadIdx.x; i < n;
       i += gridDim.x * blockDim.x)
    d[i] = (_Float16)s[i];
}

// h0 = x + positional_encoding(pos); freq[j] = (1e-4)^(j/127)  -> f16
__global__ void posenc_kernel(const float* __restrict__ x,
                              const float* __restrict__ pos,
                              _Float16* __restrict__ h16, int total)
{
  for (int i = blockIdx.x * blockDim.x + threadIdx.x; i < total;
       i += gridDim.x * blockDim.x) {
    int n = i >> 8;
    int f = i & (HDIM - 1);
    float t = (float)(f & 127) * (1.0f / 127.0f);
    float freq = expf(-9.210340371976184f * t);   // ln(1e-4) * t
    float ang = pos[n] * freq;
    float pe = (f < 128) ? sinf(ang) : cosf(ang);
    h16[i] = (_Float16)(x[i] + pe);
  }
}

__global__ void zero_i32_kernel(int* __restrict__ p, int n) {
  int i = blockIdx.x * blockDim.x + threadIdx.x;
  if (i < n) p[i] = 0;
}

__global__ void degree_kernel(const int* __restrict__ dst, int* __restrict__ deg, int e) {
  int i = blockIdx.x * blockDim.x + threadIdx.x;
  if (i < e) atomicAdd(&deg[dst[i]], 1);
}

// Single-block exclusive scan of deg -> offsets/cursor, plus inv_deg.
__global__ void scan_kernel(const int* __restrict__ deg,
                            int* __restrict__ offsets, int* __restrict__ cursor,
                            float* __restrict__ inv_deg, int n)
{
  __shared__ int sh[256];
  __shared__ int carrySh;
  const int t = threadIdx.x;
  if (t == 0) carrySh = 0;
  __syncthreads();
  for (int base = 0; base < n; base += 256) {
    int i = base + t;
    int v = (i < n) ? deg[i] : 0;
    sh[t] = v;
    __syncthreads();
    for (int off = 1; off < 256; off <<= 1) {
      int add = (t >= off) ? sh[t - off] : 0;
      __syncthreads();
      sh[t] += add;
      __syncthreads();
    }
    int incl = sh[t];
    int carry = carrySh;
    if (i < n) {
      int excl = carry + incl - v;
      offsets[i] = excl;
      cursor[i]  = excl;
      inv_deg[i] = (v > 0) ? 1.0f / (float)v : 0.0f;
    }
    __syncthreads();
    if (t == 255) carrySh = carry + incl;
    __syncthreads();
  }
  if (t == 0) offsets[n] = carrySh;
}

__global__ void csr_fill_kernel(const int* __restrict__ src,
                                const int* __restrict__ dst,
                                int* __restrict__ cursor,
                                int* __restrict__ csr_src, int e)
{
  int i = blockIdx.x * blockDim.x + threadIdx.x;
  if (i < e) {
    int p = atomicAdd(&cursor[dst[i]], 1);
    csr_src[p] = src[i];
  }
}

// Mean aggregation by gather: block = node, thread = feature. f16 in/out,
// f32 accumulation.
__global__ void aggregate_kernel(const _Float16* __restrict__ xp16,
                                 const int* __restrict__ csr_src,
                                 const int* __restrict__ offsets,
                                 const float* __restrict__ inv_deg,
                                 _Float16* __restrict__ agg16)
{
  const int n = blockIdx.x;
  const int f = threadIdx.x;
  const int b = offsets[n], e = offsets[n + 1];
  float acc = 0.f;
  for (int j = b; j < e; ++j) {
    int s = csr_src[j];
    acc += (float)xp16[(size_t)s * HDIM + f];
  }
  agg16[(size_t)n * HDIM + f] = (_Float16)(acc * inv_deg[n]);
}

// ------------------------------ layernorm ----------------------------------

__global__ void reduce_kernel(const float* __restrict__ h2,
                              float* __restrict__ partials, int total)
{
  __shared__ float s1[256], s2[256];
  float a = 0.f, b = 0.f;
  for (int i = blockIdx.x * 256 + threadIdx.x; i < total; i += gridDim.x * 256) {
    float v = h2[i];
    a += v; b += v * v;
  }
  s1[threadIdx.x] = a; s2[threadIdx.x] = b;
  __syncthreads();
  for (int off = 128; off > 0; off >>= 1) {
    if (threadIdx.x < off) {
      s1[threadIdx.x] += s1[threadIdx.x + off];
      s2[threadIdx.x] += s2[threadIdx.x + off];
    }
    __syncthreads();
  }
  if (threadIdx.x == 0) {
    partials[blockIdx.x]             = s1[0];
    partials[gridDim.x + blockIdx.x] = s2[0];
  }
}

__global__ void stats_kernel(const float* __restrict__ partials,
                             float* __restrict__ stats, int nparts, float invTotal)
{
  __shared__ float s1[256], s2[256];
  float a = 0.f, b = 0.f;
  for (int i = threadIdx.x; i < nparts; i += 256) {
    a += partials[i];
    b += partials[nparts + i];
  }
  s1[threadIdx.x] = a; s2[threadIdx.x] = b;
  __syncthreads();
  for (int off = 128; off > 0; off >>= 1) {
    if (threadIdx.x < off) {
      s1[threadIdx.x] += s1[threadIdx.x + off];
      s2[threadIdx.x] += s2[threadIdx.x + off];
    }
    __syncthreads();
  }
  if (threadIdx.x == 0) {
    float mu  = s1[0] * invTotal;
    float var = s2[0] * invTotal - mu * mu;
    stats[0] = mu;
    stats[1] = rsqrtf(var + LN_EPS);
  }
}

// h16 = f16( leaky_relu((h2 - mu) * rstd * w[f] + b[f], 0.2) )
__global__ void ln_leaky_kernel(const float* __restrict__ h2,
                                const float* __restrict__ stats,
                                const float* __restrict__ w,
                                const float* __restrict__ b,
                                _Float16* __restrict__ h16, int total)
{
  const float mu = stats[0], rs = stats[1];
  for (int i = blockIdx.x * blockDim.x + threadIdx.x; i < total;
       i += gridDim.x * blockDim.x) {
    int f = i & (HDIM - 1);
    float v = (h2[i] - mu) * rs * w[f] + b[f];
    h16[i] = (_Float16)(v > 0.f ? v : 0.2f * v);
  }
}

// ---------------------------------------------------------------------------

static inline size_t align_up(size_t v, size_t a) { return (v + a - 1) & ~(a - 1); }

extern "C" void kernel_launch(void* const* d_in, const int* in_sizes, int n_in,
                              void* d_out, int out_size, void* d_ws, size_t ws_size,
                              hipStream_t stream)
{
  const float* x     = (const float*)d_in[0];
  const float* pos   = (const float*)d_in[1];
  const int*   ei    = (const int*)  d_in[2];
  const float* Wp    = (const float*)d_in[3];
  const float* bp    = (const float*)d_in[4];
  const float* Wl    = (const float*)d_in[5];
  const float* bl    = (const float*)d_in[6];
  const float* Wr    = (const float*)d_in[7];
  const float* ln_w  = (const float*)d_in[8];
  const float* ln_b  = (const float*)d_in[9];
  const float* W_out = (const float*)d_in[10];
  const float* b_out = (const float*)d_in[11];

  const int N = in_sizes[1];             // 50000
  const int E = in_sizes[2] / 2;         // 500000
  const int DEPTH = in_sizes[4] / HDIM;  // 3
  const int total = N * HDIM;            // 12.8M elements
  const int WSZ = HDIM * HDIM;           // 65536
  const int* src = ei;
  const int* dst = ei + E;

  // ---- workspace carve-out (~135 MB) ----
  char* p = (char*)d_ws;
  _Float16* h16   = (_Float16*)p; p += align_up((size_t)total * 2, 256);
  _Float16* xp16  = (_Float16*)p; p += align_up((size_t)total * 2, 256);
  _Float16* agg16 = (_Float16*)p; p += align_up((size_t)total * 2, 256);
  float*    h2Buf = (float*)p;    p += align_up((size_t)total * 4, 256);
  _Float16* Wp16  = (_Float16*)p; p += align_up((size_t)DEPTH * WSZ * 2, 256);
  _Float16* Wl16  = (_Float16*)p; p += align_up((size_t)DEPTH * WSZ * 2, 256);
  _Float16* Wr16  = (_Float16*)p; p += align_up((size_t)DEPTH * WSZ * 2, 256);
  _Float16* Wo16  = (_Float16*)p; p += align_up((size_t)WSZ * 2, 256);
  int*   deg    = (int*)p;   p += align_up((size_t)N * 4, 256);
  float* invDeg = (float*)p; p += align_up((size_t)N * 4, 256);
  int*   offs   = (int*)p;   p += align_up((size_t)(N + 1) * 4, 256);
  int*   cursor = (int*)p;   p += align_up((size_t)N * 4, 256);
  int*   csr    = (int*)p;   p += align_up((size_t)E * 4, 256);
  float* parts  = (float*)p; p += align_up((size_t)2 * 1024 * 4, 256);
  float* stats  = (float*)p; p += align_up((size_t)2 * 4, 256);

  const int rowTiles = (N + 31) / 32;  // 1563
  const int NBLK = 2048;               // grid-stride elementwise grids
  const int RBLK = 1024;               // reduction partial blocks

  // One-time f32->f16 weight conversion (10 x 64K elements, negligible)
  cvt_f32_to_f16_kernel<<<256, 256, 0, stream>>>(Wp,    Wp16, DEPTH * WSZ);
  cvt_f32_to_f16_kernel<<<256, 256, 0, stream>>>(Wl,    Wl16, DEPTH * WSZ);
  cvt_f32_to_f16_kernel<<<256, 256, 0, stream>>>(Wr,    Wr16, DEPTH * WSZ);
  cvt_f32_to_f16_kernel<<<256, 256, 0, stream>>>(W_out, Wo16, WSZ);

  // h0 = x + posenc(pos)  (f16)
  posenc_kernel<<<NBLK, 256, 0, stream>>>(x, pos, h16, total);

  // CSR build (once per call; int atomics only)
  zero_i32_kernel<<<(N + 255) / 256, 256, 0, stream>>>(deg, N);
  degree_kernel<<<(E + 255) / 256, 256, 0, stream>>>(dst, deg, E);
  scan_kernel<<<1, 256, 0, stream>>>(deg, offs, cursor, invDeg, N);
  csr_fill_kernel<<<(E + 255) / 256, 256, 0, stream>>>(src, dst, cursor, csr, E);

  for (int l = 0; l < DEPTH && l < 8; ++l) {
    const _Float16* Wpl = Wp16 + (size_t)l * WSZ;
    const _Float16* Wll = Wl16 + (size_t)l * WSZ;
    const _Float16* Wrl = Wr16 + (size_t)l * WSZ;
    const float* bpl = bp + (size_t)l * HDIM;
    const float* bll = bl + (size_t)l * HDIM;
    const float* lwl = ln_w + (size_t)l * HDIM;
    const float* lbl = ln_b + (size_t)l * HDIM;

    // xp16 = f16(relu(h @ Wp^T + bp))
    gemm_wmma_kernel<0><<<rowTiles, 128, 0, stream>>>(
        h16, Wpl, nullptr, nullptr, bpl, nullptr, (void*)xp16, N);
    // agg16 = mean-gather of xp over in-neighbors (f32 accumulation)
    aggregate_kernel<<<N, HDIM, 0, stream>>>(xp16, csr, offs, invDeg, agg16);
    // h2 = agg @ Wl^T + h @ Wr^T + bl  (f32, for exact LN stats)
    gemm_wmma_kernel<1><<<rowTiles, 128, 0, stream>>>(
        agg16, Wll, h16, Wrl, bll, nullptr, (void*)h2Buf, N);
    // graph layernorm stats + fused normalize/leaky_relu -> h16
    reduce_kernel<<<RBLK, 256, 0, stream>>>(h2Buf, parts, total);
    stats_kernel<<<1, 256, 0, stream>>>(parts, stats, RBLK, 1.0f / (float)total);
    ln_leaky_kernel<<<NBLK, 256, 0, stream>>>(h2Buf, stats, lwl, lbl, h16, total);
  }

  // out = x + h @ W_out^T + b_out   (f32)
  gemm_wmma_kernel<2><<<rowTiles, 128, 0, stream>>>(
      h16, Wo16, nullptr, nullptr, b_out, x, d_out, N);
}
